// NexusNet_85358180041622
// MI455X (gfx1250) — matrix operand order repaired and checked by reading, not compiled, not run on hardware
//
#include <hip/hip_runtime.h>

#define CC 5     // classes
#define FF 64    // node feature width (also S = EF = 64)
#define PP 3     // planes
#define KNEX 192 // P*F nexus layer-1 K
#define KCAT 128 // F+S edge/node layer-1 K

typedef __attribute__((ext_vector_type(16))) __bf16 v16bf;
typedef __attribute__((ext_vector_type(8)))  __bf16 v8bf;
typedef __attribute__((ext_vector_type(8)))  float  v8f;

// ---------------------------------------------------------------- helpers ---

__device__ __forceinline__ unsigned short bfbits(float f) {
  union { float f; unsigned u; } a; a.f = f;
  return (unsigned short)((a.u + 0x7FFFu + ((a.u >> 16) & 1u)) >> 16);   // RNE
}

__device__ __forceinline__ __bf16 f2bf(float f) {
  union { unsigned short s; __bf16 b; } o; o.s = bfbits(f);
  return o.b;
}

__device__ __forceinline__ unsigned pack2(float lo, float hi) {
  return (unsigned)bfbits(lo) | ((unsigned)bfbits(hi) << 16);
}

// pack float4 -> 4 bf16, store as one 8-byte LDS store
__device__ __forceinline__ void store4bf(__bf16* dst, float4 v) {
  unsigned long long q = (unsigned long long)pack2(v.x, v.y) |
                         ((unsigned long long)pack2(v.z, v.w) << 32);
  *(unsigned long long*)dst = q;
}

__device__ __forceinline__ v8f v8zero() {
  v8f v;
#pragma unroll
  for (int i = 0; i < 8; ++i) v[i] = 0.0f;
  return v;
}

// A fragment: 16x32 bf16 tile (MxK), row-major in LDS, leading dim lda.
// lanes 0-15 -> K 0..7 / 16..23, lanes 16-31 -> K 8..15 / 24..31.
// Two 16B contiguous LDS loads per lane.
__device__ __forceinline__ v16bf load_a_frag(const __bf16* p, int lda, int lane) {
  const int row = lane & 15;
  const int kb  = (lane >> 4) << 3;
  const __bf16* q = p + row * lda + kb;
  v8bf lo = *(const v8bf*)q;
  v8bf hi = *(const v8bf*)(q + 16);
  v16bf a;
#pragma unroll
  for (int t = 0; t < 8; ++t) { a[t] = lo[t]; a[t + 8] = hi[t]; }
  return a;
}

// B fragment from pre-swizzled panel: layout [kk][nt][lane][16] contiguous.
__device__ __forceinline__ v16bf load_b_swz(const __bf16* sW, int kk, int nt, int lane) {
  const __bf16* q = sW + ((((kk << 2) + nt) << 5) + lane) * 16;
  v8bf lo = *(const v8bf*)q;
  v8bf hi = *(const v8bf*)(q + 8);
  v16bf b;
#pragma unroll
  for (int t = 0; t < 8; ++t) { b[t] = lo[t]; b[t + 8] = hi[t]; }
  return b;
}

__device__ __forceinline__ v8f wmma_bf16(v16bf a, v16bf b, v8f c) {
  return __builtin_amdgcn_wmma_f32_16x16x32_bf16(
      /*neg_a=*/false, a, /*neg_b=*/false, b,
      /*c_mod=*/(short)0, c, /*reuse_a=*/false, /*reuse_b=*/false);
}

__device__ __forceinline__ void wave_lds_fence() {
  asm volatile("s_wait_dscnt 0" ::: "memory");
}

// ---------------------------------------------------------------- kernels ---

__global__ void k_zero(float* __restrict__ p, int n) {
  int i = blockIdx.x * blockDim.x + threadIdx.x;
  if (i < n) p[i] = 0.0f;
}

// convert f32 weight panels [panels][K][64] into bf16 fragment-major layout:
// dst[pnl][kk][nt][lane][t] = src[pnl][kk*32 + (lane>=16)*8 + t + (t>=8)*8][nt*16 + (lane&15)]
__global__ void k_cvt_swz(const float* __restrict__ src, __bf16* __restrict__ dst,
                          int panels, int ksteps) {
  int per = ksteps << 11;                  // ksteps*2048 elems per panel
  int total = panels * per;
  int idx = blockIdx.x * blockDim.x + threadIdx.x;
  if (idx >= total) return;
  int pnl = idx / per;
  int r   = idx - pnl * per;
  int t    = r & 15;
  int lane = (r >> 4) & 31;
  int nt   = (r >> 9) & 3;
  int kk   = r >> 11;
  int row = (kk << 5) + ((lane >> 4) << 3) + t + ((t >> 3) << 3);
  int col = (nt << 4) + (lane & 15);
  dst[idx] = f2bf(src[(size_t)pnl * (ksteps << 11) + row * 64 + col]);
}

// segment_sum(x[src] -> nconcat[dst, :, plane*64 + k]) via fp32 atomics
__global__ void k_scatter(const float* __restrict__ x, const int* __restrict__ edges,
                          int E, float* __restrict__ nconcat, int plane) {
  int idx = blockIdx.x * blockDim.x + threadIdx.x;
  int total = E * CC * FF;
  if (idx >= total) return;
  int e = idx / (CC * FF);
  int r = idx - e * CC * FF;
  int c = r >> 6;
  int k = r & 63;
  int s = edges[e];
  int d = edges[E + e];
  float v = x[((size_t)s * CC + c) * FF + k];
  atomicAdd(&nconcat[((size_t)d * CC + c) * KNEX + plane * FF + k], v);
}

// nexus MLP: [M,C,192] -> tanh -> [M,C,64] -> tanh -> [M,C,64]
__global__ void __launch_bounds__(128) k_nexus_mlp(
    const float* __restrict__ nconcat,
    const __bf16* __restrict__ w1, const float* __restrict__ b1,
    const __bf16* __restrict__ w2, const float* __restrict__ b2,
    float* __restrict__ nout) {
  const int c    = blockIdx.y;
  const int lane = threadIdx.x & 31;
  const int wave = threadIdx.x >> 5;
  const int row0 = (blockIdx.x * 4 + wave) * 16;

  __shared__ __attribute__((aligned(16))) __bf16 sW1[KNEX * 64];   // 24 KB
  __shared__ __attribute__((aligned(16))) __bf16 sW2[64 * 64];     //  8 KB
  __shared__ __attribute__((aligned(16))) __bf16 sA[4][16 * KNEX]; // 24 KB

  { // block-cooperative 16B weight copies
    const uint4* g1 = (const uint4*)(w1 + (size_t)c * KNEX * 64);
    const uint4* g2 = (const uint4*)(w2 + (size_t)c * 64 * 64);
    uint4* s1 = (uint4*)sW1;
    uint4* s2 = (uint4*)sW2;
    for (int i = threadIdx.x; i < KNEX * 64 / 8; i += blockDim.x) s1[i] = g1[i];
    for (int i = threadIdx.x; i < 64 * 64 / 8;  i += blockDim.x) s2[i] = g2[i];
  }
  // stage A tile: float4 global load -> packed bf16 b64 LDS store
  for (int i = lane; i < 16 * KNEX / 4; i += 32) {
    int r = i / (KNEX / 4), kq = i - r * (KNEX / 4);
    float4 v = *(const float4*)&nconcat[((size_t)(row0 + r) * CC + c) * KNEX + kq * 4];
    store4bf(&sA[wave][r * KNEX + kq * 4], v);
  }
  __syncthreads();

  v8f acc[4];
#pragma unroll
  for (int nt = 0; nt < 4; ++nt) acc[nt] = v8zero();
#pragma unroll
  for (int kk = 0; kk < 6; ++kk) {
    v16bf af = load_a_frag(&sA[wave][kk * 32], KNEX, lane);
#pragma unroll
    for (int nt = 0; nt < 4; ++nt)
      acc[nt] = wmma_bf16(af, load_b_swz(sW1, kk, nt, lane), acc[nt]);
  }
  const int col = lane & 15;
  const int mb  = (lane < 16) ? 0 : 8;
#pragma unroll
  for (int nt = 0; nt < 4; ++nt) {
    int Nc = nt * 16 + col;
    float bias = b1[c * 64 + Nc];
#pragma unroll
    for (int r = 0; r < 8; ++r)
      sA[wave][(mb + r) * 64 + Nc] = f2bf(tanhf(acc[nt][r] + bias));
  }
  wave_lds_fence();

  v8f a2[4];
#pragma unroll
  for (int nt = 0; nt < 4; ++nt) a2[nt] = v8zero();
#pragma unroll
  for (int kk = 0; kk < 2; ++kk) {
    v16bf af = load_a_frag(&sA[wave][kk * 32], 64, lane);
#pragma unroll
    for (int nt = 0; nt < 4; ++nt)
      a2[nt] = wmma_bf16(af, load_b_swz(sW2, kk, nt, lane), a2[nt]);
  }
#pragma unroll
  for (int nt = 0; nt < 4; ++nt) {
    int Nc = nt * 16 + col;
    float bias = b2[c * 64 + Nc];
#pragma unroll
    for (int r = 0; r < 8; ++r)
      nout[((size_t)(row0 + mb + r) * CC + c) * 64 + Nc] = tanhf(a2[nt][r] + bias);
  }
}

// edge attention: h = tanh(W1 [x_i; n_j] + b1); logit = h.w2 + b2;
// softmax over classes; scatter w*n_j into agg (mean via cnt).
__global__ void __launch_bounds__(128) k_edge_mlp(
    const float* __restrict__ x, const float* __restrict__ nbuf,
    const int* __restrict__ edges, int E,
    const __bf16* __restrict__ w1, const float* __restrict__ b1,
    const float* __restrict__ w2v, const float* __restrict__ b2,
    float* __restrict__ agg, float* __restrict__ cnt) {
  const int lane = threadIdx.x & 31;
  const int wave = threadIdx.x >> 5;
  const int e0   = (blockIdx.x * 4 + wave) * 16;
  const int* eSrc = edges;
  const int* eDst = edges + E;

  __shared__ __attribute__((aligned(16))) __bf16 sW[KCAT * 64];    // 16 KB
  __shared__ __attribute__((aligned(16))) __bf16 sA[4][16 * KCAT]; // 16 KB
  __shared__ float sLg[4][CC][16];
  __shared__ float sSw[4][CC][16];

  const int col = lane & 15;
  const int mb  = (lane < 16) ? 0 : 8;

  for (int c = 0; c < CC; ++c) {
    __syncthreads();
    {
      const uint4* g = (const uint4*)(w1 + (size_t)c * KCAT * 64);
      uint4* s = (uint4*)sW;
      for (int i = threadIdx.x; i < KCAT * 64 / 8; i += blockDim.x) s[i] = g[i];
    }
    for (int i = lane; i < 16 * KCAT / 4; i += 32) {
      int e = i >> 5, kq = i & 31;        // 32 float4 chunks per edge row
      int k = kq * 4;
      float4 v;
      if (k < 64) v = *(const float4*)&x[((size_t)eSrc[e0 + e] * CC + c) * 64 + k];
      else        v = *(const float4*)&nbuf[((size_t)eDst[e0 + e] * CC + c) * 64 + (k - 64)];
      store4bf(&sA[wave][e * KCAT + k], v);
    }
    __syncthreads();

    float pl[8];
#pragma unroll
    for (int r = 0; r < 8; ++r) pl[r] = 0.0f;
#pragma unroll
    for (int nt = 0; nt < 4; ++nt) {
      v8f acc = v8zero();
#pragma unroll
      for (int kk = 0; kk < 4; ++kk) {
        v16bf af = load_a_frag(&sA[wave][kk * 32], KCAT, lane);
        acc = wmma_bf16(af, load_b_swz(sW, kk, nt, lane), acc);
      }
      int Nc = nt * 16 + col;
      float bias = b1[c * 64 + Nc];
      float wc   = w2v[c * 64 + Nc];
#pragma unroll
      for (int r = 0; r < 8; ++r) pl[r] += tanhf(acc[r] + bias) * wc;
    }
    // reduce across the 16 N-lanes of each half (rows mb..mb+7)
#pragma unroll
    for (int m = 1; m < 16; m <<= 1) {
#pragma unroll
      for (int r = 0; r < 8; ++r) pl[r] += __shfl_xor(pl[r], m, 32);
    }
    if (col == 0) {
      float bb = b2[c];
#pragma unroll
      for (int r = 0; r < 8; ++r) sLg[wave][c][mb + r] = pl[r] + bb;
    }
  }
  __syncthreads();

  if (lane < 16) {
    float mx = sLg[wave][0][lane];
#pragma unroll
    for (int c = 1; c < CC; ++c) mx = fmaxf(mx, sLg[wave][c][lane]);
    float ex[CC], sum = 0.0f;
#pragma unroll
    for (int c = 0; c < CC; ++c) { ex[c] = __expf(sLg[wave][c][lane] - mx); sum += ex[c]; }
    float inv = 1.0f / sum;
#pragma unroll
    for (int c = 0; c < CC; ++c) sSw[wave][c][lane] = ex[c] * inv;
    atomicAdd(&cnt[eSrc[e0 + lane]], 1.0f);
  }
  __syncthreads();

  for (int i = lane; i < 16 * 64; i += 32) {
    int e = i >> 6, k = i & 63;
    int sN = eSrc[e0 + e], dN = eDst[e0 + e];
#pragma unroll
    for (int c = 0; c < CC; ++c) {
      float val = sSw[wave][c][e] * nbuf[((size_t)dN * CC + c) * 64 + k];
      atomicAdd(&agg[((size_t)sN * CC + c) * 64 + k], val);
    }
  }
}

// node MLP: u = tanh(W2 tanh(W1 [x; agg/cnt] + b1) + b2) -> out[p]
__global__ void __launch_bounds__(128) k_node_mlp(
    const float* __restrict__ x, const float* __restrict__ agg,
    const float* __restrict__ cnt,
    const __bf16* __restrict__ w1, const float* __restrict__ b1,
    const __bf16* __restrict__ w2, const float* __restrict__ b2,
    float* __restrict__ out) {
  const int c    = blockIdx.y;
  const int lane = threadIdx.x & 31;
  const int wave = threadIdx.x >> 5;
  const int row0 = (blockIdx.x * 4 + wave) * 16;

  __shared__ __attribute__((aligned(16))) __bf16 sW1[KCAT * 64];   // 16 KB
  __shared__ __attribute__((aligned(16))) __bf16 sW2[64 * 64];     //  8 KB
  __shared__ __attribute__((aligned(16))) __bf16 sA[4][16 * KCAT]; // 16 KB

  {
    const uint4* g1 = (const uint4*)(w1 + (size_t)c * KCAT * 64);
    const uint4* g2 = (const uint4*)(w2 + (size_t)c * 64 * 64);
    uint4* s1 = (uint4*)sW1;
    uint4* s2 = (uint4*)sW2;
    for (int i = threadIdx.x; i < KCAT * 64 / 8; i += blockDim.x) s1[i] = g1[i];
    for (int i = threadIdx.x; i < 64 * 64 / 8;  i += blockDim.x) s2[i] = g2[i];
  }
  for (int i = lane; i < 16 * KCAT / 4; i += 32) {
    int r = i >> 5, kq = i & 31;
    int k = kq * 4;
    int node = row0 + r;
    float4 v;
    if (k < 64) {
      v = *(const float4*)&x[((size_t)node * CC + c) * 64 + k];
    } else {
      v = *(const float4*)&agg[((size_t)node * CC + c) * 64 + (k - 64)];
      float inv = 1.0f / fmaxf(cnt[node], 1.0f);
      v.x *= inv; v.y *= inv; v.z *= inv; v.w *= inv;
    }
    store4bf(&sA[wave][r * KCAT + k], v);
  }
  __syncthreads();

  v8f acc[4];
#pragma unroll
  for (int nt = 0; nt < 4; ++nt) acc[nt] = v8zero();
#pragma unroll
  for (int kk = 0; kk < 4; ++kk) {
    v16bf af = load_a_frag(&sA[wave][kk * 32], KCAT, lane);
#pragma unroll
    for (int nt = 0; nt < 4; ++nt)
      acc[nt] = wmma_bf16(af, load_b_swz(sW1, kk, nt, lane), acc[nt]);
  }
  const int col = lane & 15;
  const int mb  = (lane < 16) ? 0 : 8;
#pragma unroll
  for (int nt = 0; nt < 4; ++nt) {
    int Nc = nt * 16 + col;
    float bias = b1[c * 64 + Nc];
#pragma unroll
    for (int r = 0; r < 8; ++r)
      sA[wave][(mb + r) * 64 + Nc] = f2bf(tanhf(acc[nt][r] + bias));
  }
  wave_lds_fence();

  v8f a2[4];
#pragma unroll
  for (int nt = 0; nt < 4; ++nt) a2[nt] = v8zero();
#pragma unroll
  for (int kk = 0; kk < 2; ++kk) {
    v16bf af = load_a_frag(&sA[wave][kk * 32], 64, lane);
#pragma unroll
    for (int nt = 0; nt < 4; ++nt)
      a2[nt] = wmma_bf16(af, load_b_swz(sW2, kk, nt, lane), a2[nt]);
  }
#pragma unroll
  for (int nt = 0; nt < 4; ++nt) {
    int Nc = nt * 16 + col;
    float bias = b2[c * 64 + Nc];
#pragma unroll
    for (int r = 0; r < 8; ++r)
      out[((size_t)(row0 + mb + r) * CC + c) * 64 + Nc] = tanhf(a2[nt][r] + bias);
  }
}

// ------------------------------------------------------------------ host ---

static inline int cdivi(long long a, long long b) { return (int)((a + b - 1) / b); }

extern "C" void kernel_launch(void* const* d_in, const int* in_sizes, int n_in,
                              void* d_out, int out_size, void* d_ws, size_t ws_size,
                              hipStream_t stream) {
  const float* xs[PP]  = {(const float*)d_in[0], (const float*)d_in[2], (const float*)d_in[4]};
  const int*   es[PP]  = {(const int*)d_in[1],   (const int*)d_in[3],   (const int*)d_in[5]};
  const int N = in_sizes[0] / (CC * FF);
  const int E = in_sizes[1] / 2;
  const int M = in_sizes[6] / (CC * FF);

  const float* nex_w1  = (const float*)d_in[7];
  const float* nex_b1  = (const float*)d_in[8];
  const float* nex_w2  = (const float*)d_in[9];
  const float* nex_b2  = (const float*)d_in[10];
  const float* edge_w1 = (const float*)d_in[11];
  const float* edge_b1 = (const float*)d_in[12];
  const float* edge_w2 = (const float*)d_in[13];
  const float* edge_b2 = (const float*)d_in[14];
  const float* node_w1 = (const float*)d_in[15];
  const float* node_b1 = (const float*)d_in[16];
  const float* node_w2 = (const float*)d_in[17];
  const float* node_b2 = (const float*)d_in[18];

  // Workspace: region A (nconcat, later reused as agg+cnt) | nbuf | bf16 weights
  char* ws = (char*)d_ws;
  float* nconcat = (float*)ws;                                 // M*C*192 f32
  float* agg     = (float*)ws;                                 // N*C*64  f32 (reuse)
  float* cnt     = (float*)(ws + (size_t)N * CC * FF * 4);     // N       f32
  size_t regA = (size_t)M * CC * KNEX * 4;
  size_t need = (size_t)N * CC * FF * 4 + (size_t)N * 4;
  if (need > regA) regA = need;
  float*  nbuf = (float*)(ws + regA);                          // M*C*64 f32
  size_t off = regA + (size_t)M * CC * FF * 4;
  __bf16* bNexW1  = (__bf16*)(ws + off); off += (size_t)CC * KNEX * 64 * 2;
  __bf16* bNexW2  = (__bf16*)(ws + off); off += (size_t)CC * 64 * 64 * 2;
  __bf16* bEdgeW1 = (__bf16*)(ws + off); off += (size_t)PP * CC * KCAT * 64 * 2;
  __bf16* bNodeW1 = (__bf16*)(ws + off); off += (size_t)PP * CC * KCAT * 64 * 2;
  __bf16* bNodeW2 = (__bf16*)(ws + off); off += (size_t)PP * CC * 64 * 64 * 2;

  // 1) zero nexus-concat accumulator; convert + fragment-swizzle weights
  k_zero<<<cdivi((long long)M * CC * KNEX, 256), 256, 0, stream>>>(nconcat, M * CC * KNEX);
  k_cvt_swz<<<cdivi(CC * KNEX * 64, 256), 256, 0, stream>>>(nex_w1, bNexW1, CC, 6);
  k_cvt_swz<<<cdivi(CC * 64 * 64, 256), 256, 0, stream>>>(nex_w2, bNexW2, CC, 2);
  k_cvt_swz<<<cdivi(PP * CC * KCAT * 64, 256), 256, 0, stream>>>(edge_w1, bEdgeW1, PP * CC, 4);
  k_cvt_swz<<<cdivi(PP * CC * KCAT * 64, 256), 256, 0, stream>>>(node_w1, bNodeW1, PP * CC, 4);
  k_cvt_swz<<<cdivi(PP * CC * 64 * 64, 256), 256, 0, stream>>>(node_w2, bNodeW2, PP * CC, 2);

  // 2) segment-sum plane features into nexus
  for (int p = 0; p < PP; ++p)
    k_scatter<<<cdivi((long long)E * CC * FF, 256), 256, 0, stream>>>(xs[p], es[p], E, nconcat, p);

  // 3) nexus MLP (WMMA bf16)
  k_nexus_mlp<<<dim3(M / 64, CC), 128, 0, stream>>>(nconcat, bNexW1, nex_b1, bNexW2, nex_b2, nbuf);

  // 4) per-plane: edge attention + mean aggregation + node MLP
  for (int p = 0; p < PP; ++p) {
    k_zero<<<cdivi((long long)N * CC * FF, 256), 256, 0, stream>>>(agg, N * CC * FF);
    k_zero<<<cdivi(N, 256), 256, 0, stream>>>(cnt, N);
    k_edge_mlp<<<E / 64, 128, 0, stream>>>(
        xs[p], nbuf, es[p], E,
        bEdgeW1 + (size_t)p * CC * KCAT * 64, edge_b1 + (size_t)p * CC * 64,
        edge_w2 + (size_t)p * CC * 64,        edge_b2 + (size_t)p * CC,
        agg, cnt);
    k_node_mlp<<<dim3(N / 64, CC), 128, 0, stream>>>(
        xs[p], agg, cnt,
        bNodeW1 + (size_t)p * CC * KCAT * 64, node_b1 + (size_t)p * CC * 64,
        bNodeW2 + (size_t)p * CC * 64 * 64,   node_b2 + (size_t)p * CC * 64,
        (float*)d_out + (size_t)p * N * CC * FF);
  }
}
